// STNkd_39324720562888
// MI455X (gfx1250) — matrix operand order repaired and checked by reading, not compiled, not run on hardware
//
#include <hip/hip_runtime.h>
#include <hip/hip_bf16.h>

// ---------------------------------------------------------------------------
// VN network on MI455X (gfx1250, wave32). All channel-contraction GEMMs run on
// v_wmma_f32_16x16x32_bf16, f32 accumulation, bf16 intermediates in HBM.
// Weights are pre-packed to zero-padded bf16 so A-fragments are loaded
// per-lane directly from global (ISA A-layout = 2 runs of 8 consecutive K per
// lane) and the hot loop has no bounds branches (clamped addressing + zero-A).
// GEMM modes:
//   0: Y = W*X                      2: d = W*X, BN-normalize q, VN-ReLU -> h
//   1: Y = W*X (=q) + norm stats    3: d2 = W*X, Y = dot(h, d2)  (pool)
// Workgroup tile: 32 rows x 64 cols x 3 vector components (4 waves).
// ---------------------------------------------------------------------------

typedef __attribute__((ext_vector_type(16))) __bf16           bf16x16;
typedef __attribute__((ext_vector_type(16))) unsigned short   u16x16;
typedef __attribute__((ext_vector_type(8)))  unsigned short   u16x8;
typedef __attribute__((ext_vector_type(8)))  float            v8f;
typedef __attribute__((ext_vector_type(4)))  float            f32x4;

#define VN_EPS 1e-6f
#define BN_EPS 1e-5f

__device__ __forceinline__ unsigned short f2bf(float f) {
    unsigned u = __float_as_uint(f);
    unsigned r = u + 0x7FFFu + ((u >> 16) & 1u);      // RNE
    return (unsigned short)(r >> 16);
}
__device__ __forceinline__ float bf2f(unsigned short h) {
    return __uint_as_float(((unsigned)h) << 16);
}

// ISA 16-bit B-matrix 32x16: lanes 0-15 = N (K 0..7,16..23), lanes 16-31 = N (K 8..15,24..31).
__device__ __forceinline__ int b_swz(int k, int n) {
    return (n + (((k >> 3) & 1) << 4)) * 16 + (k & 7) + ((k >> 4) << 3);
}

// f32 [M,K] -> bf16 [Mp,Kp], zero padded
__global__ void pack_w(const float* __restrict__ src, unsigned short* __restrict__ dst,
                       int M, int K, int Kp, int total)
{
    int i = blockIdx.x * 256 + threadIdx.x;
    if (i >= total) return;
    int m = i / Kp, k = i % Kp;
    float v = (m < M && k < K) ? src[(long)m * K + k] : 0.f;
    dst[i] = f2bf(v);
}

template <int MODE, int XB, int QB, int YB>
__global__ __launch_bounds__(128) void vn_gemm(
    const unsigned short* __restrict__ Wb,  // [Mp,Kp] bf16, zero padded
    const void*  __restrict__ Xv,           // [B,K,3,N]  f32 or bf16
    const void*  __restrict__ Qv,           // [B,M,3,N]  (modes 2,3)
    const float* __restrict__ sc,           // [M] (mode 2)
    const float* __restrict__ sh,           // [M] (mode 2)
    void*        __restrict__ Yv,           // [B,M,3,N] (0,1,2) or [B,M,N] f32 (3)
    float*       __restrict__ sums,         // [2*M] (mode 1)
    int M, int K, int Kp, int N)
{
    __shared__ __align__(32) unsigned short lB[3][4][512];  // v, nfrag, 32x16 swz
    __shared__ float sSum[32], sSq[32];

    const float*          Xf = (const float*)Xv;
    const unsigned short* Xh = (const unsigned short*)Xv;
    const float*          Qf = (const float*)Qv;
    const unsigned short* Qh = (const unsigned short*)Qv;
    float*                Yf = (float*)Yv;
    unsigned short*       Yh = (unsigned short*)Yv;

    const int t    = threadIdx.x;
    const int wave = t >> 5;
    const int lane = t & 31;
    const int hf   = lane >> 4;
    const int ln   = lane & 15;
    const int b    = blockIdx.z;
    const int m0   = blockIdx.y * 32;
    const int n0   = blockIdx.x * 64;

    if (MODE == 1 && t < 32) { sSum[t] = 0.f; sSq[t] = 0.f; }

    const long xbase = (long)b * K * 3 * N;
    const bool fullN = (n0 + 64 <= N);

    v8f acc[2][3];
    for (int f = 0; f < 2; ++f)
        for (int vv = 0; vv < 3; ++vv)
            for (int i = 0; i < 8; ++i) acc[f][vv][i] = 0.f;

    // per-lane A row bases (two 16-row fragments)
    const unsigned short* wrow0 = Wb + (long)(m0 + ln) * Kp + hf * 8;
    const unsigned short* wrow1 = wrow0 + (long)16 * Kp;

    for (int k0 = 0; k0 < K; k0 += 32) {
        __syncthreads();
        // speculative prefetch of next k-tile of X (global_prefetch_b8)
        if (k0 + 32 < K && t < 32) {
            int pk = k0 + 32 + t; pk = pk < K ? pk : K - 1;
            if (XB) __builtin_prefetch(&Xh[xbase + ((long)pk * 3) * N + n0], 0, 1);
            else    __builtin_prefetch(&Xf[xbase + ((long)pk * 3) * N + n0], 0, 1);
        }

        // ---- stage B tile (3*32*64 bf16, clamped branch-free) ----
        if (fullN) {
            if (XB) {
                for (int e = 0; e < 6; ++e) {          // 16B loads, 8 consecutive n
                    int idx = t + e * 128;
                    int n = (idx & 7) * 8;
                    int rest = idx >> 3;
                    int k = rest & 31, vv = rest >> 5;
                    int gk = k0 + k; gk = gk < K ? gk : K - 1;
                    u16x8 val = *(const u16x8*)&Xh[xbase + ((long)gk * 3 + vv) * N + n0 + n];
                    unsigned short* dst = &lB[vv][n >> 4][b_swz(k, n & 15)];
                    for (int j = 0; j < 8; ++j) dst[16 * j] = val[j];
                }
            } else {
                for (int e = 0; e < 12; ++e) {         // 16B loads, 4 consecutive n
                    int idx = t + e * 128;
                    int n = (idx & 15) * 4;
                    int rest = idx >> 4;
                    int k = rest & 31, vv = rest >> 5;
                    int gk = k0 + k; gk = gk < K ? gk : K - 1;
                    f32x4 val = *(const f32x4*)&Xf[xbase + ((long)gk * 3 + vv) * N + n0 + n];
                    unsigned short* dst = &lB[vv][n >> 4][b_swz(k, n & 15)];
                    for (int j = 0; j < 4; ++j) dst[16 * j] = f2bf(val[j]);
                }
            }
        } else {                                       // small-N path, clamped
            for (int e = 0; e < 48; ++e) {
                int idx = t + e * 128;
                int n = idx & 63;
                int rest = idx >> 6;
                int k = rest & 31, vv = rest >> 5;
                int gk = k0 + k; gk = gk < K ? gk : K - 1;
                int gn = n0 + n; gn = gn < N ? gn : N - 1;
                long gi = xbase + ((long)gk * 3 + vv) * N + gn;
                lB[vv][n >> 4][b_swz(k, n & 15)] = XB ? Xh[gi] : f2bf(Xf[gi]);
            }
        }
        __syncthreads();

        // ---- A fragments: direct per-lane global loads (2 x 16B each) ----
        union { u16x16 w; u16x8 h[2]; } a0u, a1u;
        a0u.h[0] = *(const u16x8*)(wrow0 + k0);
        a0u.h[1] = *(const u16x8*)(wrow0 + k0 + 16);
        a1u.h[0] = *(const u16x8*)(wrow1 + k0);
        a1u.h[1] = *(const u16x8*)(wrow1 + k0 + 16);
        bf16x16 a0 = __builtin_bit_cast(bf16x16, a0u.w);
        bf16x16 a1 = __builtin_bit_cast(bf16x16, a1u.w);

        for (int vv = 0; vv < 3; ++vv) {
            bf16x16 bm = __builtin_bit_cast(bf16x16, *(const u16x16*)&lB[vv][wave][lane * 16]);
            acc[0][vv] = __builtin_amdgcn_wmma_f32_16x16x32_bf16(
                false, a0, false, bm, (short)0, acc[0][vv], false, false);
            acc[1][vv] = __builtin_amdgcn_wmma_f32_16x16x32_bf16(
                false, a1, false, bm, (short)0, acc[1][vv], false, false);
        }
    }

    // ---------------- epilogue ----------------
    const int  gn    = n0 + wave * 16 + ln;
    const long ybase = (long)b * M * 3 * N;

    for (int f = 0; f < 2; ++f) {
        for (int r = 0; r < 8; ++r) {
            const int  gm    = m0 + f * 16 + r + 8 * hf;   // C/D: VGPR r -> M=r(+8)
            const bool valid = (gm < M) && (gn < N);

            if (MODE == 0 || MODE == 1) {
                if (valid) {
                    for (int vv = 0; vv < 3; ++vv) {
                        long yi = ybase + ((long)gm * 3 + vv) * N + gn;
                        if (YB) Yh[yi] = f2bf(acc[f][vv][r]);
                        else    Yf[yi] = acc[f][vv][r];
                    }
                }
                if (MODE == 1) {
                    float nv = 0.f;
                    if (valid)
                        nv = sqrtf(acc[f][0][r]*acc[f][0][r] + acc[f][1][r]*acc[f][1][r] +
                                   acc[f][2][r]*acc[f][2][r]) + VN_EPS;
                    float ns = nv * nv;
                    for (int off = 1; off < 16; off <<= 1) {   // reduce 16 lanes (same m)
                        nv += __shfl_xor(nv, off, 32);
                        ns += __shfl_xor(ns, off, 32);
                    }
                    if (ln == 0) {                             // lanes 0 and 16
                        atomicAdd(&sSum[f * 16 + r + 8 * hf], nv);   // ds_add_f32
                        atomicAdd(&sSq [f * 16 + r + 8 * hf], ns);
                    }
                }
            } else if (MODE == 2) {
                if (valid) {
                    long qb = ybase + (long)gm * 3 * N + gn;
                    float q0, q1, q2;
                    if (QB) { q0 = bf2f(Qh[qb]); q1 = bf2f(Qh[qb + N]); q2 = bf2f(Qh[qb + 2L*N]); }
                    else    { q0 = Qf[qb];       q1 = Qf[qb + N];       q2 = Qf[qb + 2L*N]; }
                    float nrm   = sqrtf(q0*q0 + q1*q1 + q2*q2) + VN_EPS;
                    float ratio = (sc[gm] * nrm + sh[gm]) / nrm;
                    float p0 = q0 * ratio, p1 = q1 * ratio, p2 = q2 * ratio;
                    float d0 = acc[f][0][r], d1 = acc[f][1][r], d2 = acc[f][2][r];
                    float dot = p0*d0 + p1*d1 + p2*d2;
                    if (dot < 0.f) {
                        float fc = dot / (d0*d0 + d1*d1 + d2*d2 + VN_EPS);
                        p0 -= fc * d0; p1 -= fc * d1; p2 -= fc * d2;
                    }
                    if (YB) { Yh[qb] = f2bf(p0); Yh[qb + N] = f2bf(p1); Yh[qb + 2L*N] = f2bf(p2); }
                    else    { Yf[qb] = p0;       Yf[qb + N] = p1;       Yf[qb + 2L*N] = p2; }
                }
            } else {                                    // MODE 3: pool dot (f32 out)
                if (valid) {
                    long qb = ybase + (long)gm * 3 * N + gn;
                    float h0, h1, h2;
                    if (QB) { h0 = bf2f(Qh[qb]); h1 = bf2f(Qh[qb + N]); h2 = bf2f(Qh[qb + 2L*N]); }
                    else    { h0 = Qf[qb];       h1 = Qf[qb + N];       h2 = Qf[qb + 2L*N]; }
                    Yf[(long)b * M * N + (long)gm * N + gn] =
                        h0*acc[f][0][r] + h1*acc[f][1][r] + h2*acc[f][2][r];
                }
            }
        }
    }

    if (MODE == 1) {
        __syncthreads();
        if (t < 32 && (m0 + t) < M) {
            atomicAdd(&sums[m0 + t],     sSum[t]);
            atomicAdd(&sums[M + m0 + t], sSq[t]);
        }
    }
}

__global__ void vn_finalize(const float* __restrict__ sums,
                            const float* __restrict__ g,
                            const float* __restrict__ bta,
                            float* __restrict__ sc, float* __restrict__ sh,
                            int M, float inv_cnt)
{
    int c = blockIdx.x * blockDim.x + threadIdx.x;
    if (c < M) {
        float mu  = sums[c] * inv_cnt;
        float var = sums[M + c] * inv_cnt - mu * mu;
        float A   = g[c] * rsqrtf(var + BN_EPS);
        sc[c] = A;
        sh[c] = bta[c] - A * mu;
    }
}

// one wave per (b,c): argmax_n dot[b,c,n]; gather bf16 h[b,c,:,argmax] -> f32
__global__ __launch_bounds__(128) void vn_argmax(
    const float* __restrict__ dot, const unsigned short* __restrict__ h,
    float* __restrict__ out, int B, int C, int N)
{
    int w    = blockIdx.x * 4 + (threadIdx.x >> 5);
    int lane = threadIdx.x & 31;
    if (w >= B * C) return;
    int b = w / C, c = w % C;
    const float* dp = dot + ((long)b * C + c) * N;
    float best = -3.402823466e38f;
    int   bi   = 0;
    for (int n = lane; n < N; n += 32) {
        float v = dp[n];
        if (v > best) { best = v; bi = n; }
    }
    for (int off = 16; off; off >>= 1) {
        float ov = __shfl_xor(best, off, 32);
        int   oi = __shfl_xor(bi,   off, 32);
        if (ov > best || (ov == best && oi < bi)) { best = ov; bi = oi; }
    }
    const unsigned short* hp = h + ((long)b * C + c) * 3 * N;
    if (lane < 3) out[((long)b * C + c) * 3 + lane] = bf2f(hp[(long)lane * N + bi]);
}

// ---------------------------------------------------------------------------

extern "C" void kernel_launch(void* const* d_in, const int* in_sizes, int n_in,
                              void* d_out, int out_size, void* d_ws, size_t ws_size,
                              hipStream_t stream)
{
    const int B = 16, N = 4096;
    const int D = 64, C1 = 21, C2 = 42, C3 = 341, C4 = 170, C5 = 85;

    const float* x   = (const float*)d_in[0];
    const float* w1f = (const float*)d_in[1];
    const float* w1d = (const float*)d_in[2];
    const float* g1  = (const float*)d_in[3];
    const float* b1  = (const float*)d_in[4];
    const float* w2f = (const float*)d_in[5];
    const float* w2d = (const float*)d_in[6];
    const float* g2  = (const float*)d_in[7];
    const float* b2  = (const float*)d_in[8];
    const float* w3f = (const float*)d_in[9];
    const float* w3d = (const float*)d_in[10];
    const float* g3  = (const float*)d_in[11];
    const float* b3  = (const float*)d_in[12];
    const float* wp  = (const float*)d_in[13];
    const float* w4f = (const float*)d_in[14];
    const float* w4d = (const float*)d_in[15];
    const float* g4  = (const float*)d_in[16];
    const float* b4  = (const float*)d_in[17];
    const float* w5f = (const float*)d_in[18];
    const float* w5d = (const float*)d_in[19];
    const float* g5  = (const float*)d_in[20];
    const float* b5  = (const float*)d_in[21];
    const float* w6  = (const float*)d_in[22];

    // ---- workspace carve-out (256B aligned) ----
    char* p = (char*)d_ws;
    auto alloc = [&](size_t bytes) { char* r = p; p += (bytes + 255) & ~(size_t)255; return r; };

    const long QE  = (long)B * C3 * 3 * N;
    const long H1E = (long)B * C1 * 3 * N;
    const long H2E = (long)B * C2 * 3 * N;

    unsigned short* Q   = (unsigned short*)alloc(QE * 2);
    unsigned short* H1  = (unsigned short*)alloc(H1E * 2);
    unsigned short* H2  = (unsigned short*)alloc(H2E * 2);
    unsigned short* H3  = (unsigned short*)alloc(QE * 2);
    float* POOL = (float*)alloc((size_t)B * C3 * 3 * 4);
    float* Q4   = (float*)alloc((size_t)B * C4 * 3 * 4);
    float* H4   = (float*)alloc((size_t)B * C4 * 3 * 4);
    float* Q5   = (float*)alloc((size_t)B * C5 * 3 * 4);
    float* H5   = (float*)alloc((size_t)B * C5 * 3 * 4);
    float* SUMS = (float*)alloc(2 * C3 * 4);
    float* SC   = (float*)alloc(C3 * 4);
    float* SH   = (float*)alloc(C3 * 4);
    float* DOT  = (float*)Q;       // alias: q3 dead before pool pass

    // packed bf16 weights (zero padded M->32k, K->32k)
    auto padM = [](int M) { return (M + 31) & ~31; };
    auto padK = [](int K) { return (K + 31) & ~31; };
    auto packW = [&](const float* src, int M, int K) {
        int Mp = padM(M), Kp = padK(K), total = Mp * Kp;
        unsigned short* dst = (unsigned short*)alloc((size_t)total * 2);
        pack_w<<<dim3((total + 255) / 256), 256, 0, stream>>>(src, dst, M, K, Kp, total);
        return dst;
    };
    const unsigned short* P1f = packW(w1f, C1, D),  *P1d = packW(w1d, C1, D);
    const unsigned short* P2f = packW(w2f, C2, C1), *P2d = packW(w2d, C2, C1);
    const unsigned short* P3f = packW(w3f, C3, C2), *P3d = packW(w3d, C3, C2);
    const unsigned short* Pp  = packW(wp,  C3, C3);
    const unsigned short* P4f = packW(w4f, C4, C3), *P4d = packW(w4d, C4, C3);
    const unsigned short* P5f = packW(w5f, C5, C4), *P5d = packW(w5d, C5, C4);
    const unsigned short* P6  = packW(w6,  D,  C5);

    float* OUT = (float*)d_out;
    const float invBN = 1.0f / (float)(B * N);
    const float invB  = 1.0f / (float)B;
    const dim3 blk(128);
    auto grid = [&](int M, int Nn) { return dim3((Nn + 63) / 64, (M + 31) / 32, 16); };

    // ---- block 1: x f32 -> H1 bf16 ----
    hipMemsetAsync(SUMS, 0, 2 * C1 * sizeof(float), stream);
    vn_gemm<1,0,0,1><<<grid(C1,N), blk, 0, stream>>>(P1f, x, nullptr, nullptr, nullptr, Q, SUMS, C1, D, padK(D), N);
    vn_finalize<<<dim3(1), 256, 0, stream>>>(SUMS, g1, b1, SC, SH, C1, invBN);
    vn_gemm<2,0,1,1><<<grid(C1,N), blk, 0, stream>>>(P1d, x, Q, SC, SH, H1, nullptr, C1, D, padK(D), N);

    // ---- block 2: H1 -> H2 ----
    hipMemsetAsync(SUMS, 0, 2 * C2 * sizeof(float), stream);
    vn_gemm<1,1,0,1><<<grid(C2,N), blk, 0, stream>>>(P2f, H1, nullptr, nullptr, nullptr, Q, SUMS, C2, C1, padK(C1), N);
    vn_finalize<<<dim3(1), 256, 0, stream>>>(SUMS, g2, b2, SC, SH, C2, invBN);
    vn_gemm<2,1,1,1><<<grid(C2,N), blk, 0, stream>>>(P2d, H1, Q, SC, SH, H2, nullptr, C2, C1, padK(C1), N);

    // ---- block 3: H2 -> H3 ----
    hipMemsetAsync(SUMS, 0, 2 * C3 * sizeof(float), stream);
    vn_gemm<1,1,0,1><<<grid(C3,N), blk, 0, stream>>>(P3f, H2, nullptr, nullptr, nullptr, Q, SUMS, C3, C2, padK(C2), N);
    vn_finalize<<<dim3((C3+255)/256), 256, 0, stream>>>(SUMS, g3, b3, SC, SH, C3, invBN);
    vn_gemm<2,1,1,1><<<grid(C3,N), blk, 0, stream>>>(P3d, H2, Q, SC, SH, H3, nullptr, C3, C2, padK(C2), N);

    // ---- max pool: dot = <h3, Wp h3>, argmax over N, gather ----
    vn_gemm<3,1,1,0><<<grid(C3,N), blk, 0, stream>>>(Pp, H3, H3, nullptr, nullptr, DOT, nullptr, C3, C3, padK(C3), N);
    vn_argmax<<<dim3((B * C3 + 3) / 4), 128, 0, stream>>>(DOT, H3, POOL, B, C3, N);

    // ---- block 4: POOL f32 [B,341,3] -> H4 f32 [B,170,3] (N=1, BN over batch) ----
    hipMemsetAsync(SUMS, 0, 2 * C4 * sizeof(float), stream);
    vn_gemm<1,0,0,0><<<grid(C4,1), blk, 0, stream>>>(P4f, POOL, nullptr, nullptr, nullptr, Q4, SUMS, C4, C3, padK(C3), 1);
    vn_finalize<<<dim3(1), 256, 0, stream>>>(SUMS, g4, b4, SC, SH, C4, invB);
    vn_gemm<2,0,0,0><<<grid(C4,1), blk, 0, stream>>>(P4d, POOL, Q4, SC, SH, H4, nullptr, C4, C3, padK(C3), 1);

    // ---- block 5: H4 -> H5 ----
    hipMemsetAsync(SUMS, 0, 2 * C5 * sizeof(float), stream);
    vn_gemm<1,0,0,0><<<grid(C5,1), blk, 0, stream>>>(P5f, H4, nullptr, nullptr, nullptr, Q5, SUMS, C5, C4, padK(C4), 1);
    vn_finalize<<<dim3(1), 256, 0, stream>>>(SUMS, g5, b5, SC, SH, C5, invB);
    vn_gemm<2,0,0,0><<<grid(C5,1), blk, 0, stream>>>(P5d, H4, Q5, SC, SH, H5, nullptr, C5, C4, padK(C4), 1);

    // ---- final linear: H5 -> OUT f32 [B,64,3] ----
    vn_gemm<0,0,0,0><<<grid(D,1), blk, 0, stream>>>(P6, H5, nullptr, nullptr, nullptr, OUT, nullptr, D, C5, padK(C5), 1);

    (void)in_sizes; (void)n_in; (void)out_size; (void)ws_size;
}
